// VectorQuantizer_62594853372132
// MI455X (gfx1250) — matrix operand order repaired and checked by reading, not compiled, not run on hardware
//
#include <hip/hip_runtime.h>
#include <hip/hip_bf16.h>

// ---------------------------------------------------------------------------
// VQ-VAE forward for MI455X (gfx1250, wave32, WMMA).
//   d = ||e||^2 - 2 z.e   (||z||^2 dropped: row-constant, argmin-invariant)
//   GEMM term in bf16 WMMA 16x16x32 w/ f32 accumulate; gather+loss in fp32.
//   B chunks double-buffered in LDS via async global->LDS loads (ASYNCcnt).
// ---------------------------------------------------------------------------

#define N_TOK   65536
#define E_DIM   256
#define N_E     1024
#define BETA    0.25f

typedef __attribute__((ext_vector_type(16))) __bf16          v16bf;
typedef __attribute__((ext_vector_type(8)))  float           v8f;
typedef __attribute__((ext_vector_type(16))) unsigned short  us16;
typedef __attribute__((ext_vector_type(4)))  float           f4;

// round-to-nearest-even f32 -> bf16
__device__ __forceinline__ unsigned short f2bf(float f) {
    unsigned u = __builtin_bit_cast(unsigned, f);
    u += 0x7FFFu + ((u >> 16) & 1u);
    return (unsigned short)(u >> 16);
}

// Async global->LDS copy of 32B (2 x b128). ISA: INST_OFFSET is added to BOTH
// the LDS address (VDST VGPR) and the global address, so one VGPR set serves
// both halves. Tracked by ASYNCcnt.
__device__ __forceinline__ void async_copy32(unsigned lds_off, unsigned long long gaddr) {
    asm volatile("global_load_async_to_lds_b128 %0, %1, off\n\t"
                 "global_load_async_to_lds_b128 %0, %1, off offset:4096"
                 :: "v"(lds_off), "v"(gaddr)
                 : "memory");
}
__device__ __forceinline__ void wait_async0() {
    asm volatile("s_wait_asynccnt 0x0" ::: "memory");
}

// ---------------------------------------------------------------------------
// Kernel 1: pre-swizzle codebook f32 -> bf16 into WMMA B-fragment order.
// Layout: chunk cb (16 codes), kstep ks (32 K), lane (32):
//   lane<16  : col = lane,    K = ks*32 + 0..15   (16 consecutive halves)
//   lane>=16 : col = lane-16, K = ks*32 + 16..31
// Linear: embq[(((cb*8)+ks)*32 + lane)*16 + h]
// ---------------------------------------------------------------------------
__launch_bounds__(256)
__global__ void vq_prep_layout_k(const float* __restrict__ emb,
                                 unsigned short* __restrict__ embq) {
    const int cb   = blockIdx.x;        // 0..63
    const int ks   = threadIdx.x >> 5;  // 0..7
    const int lane = threadIdx.x & 31;
    const int col  = lane & 15;
    const int kb   = ks * 32 + ((lane >= 16) ? 16 : 0);
    const float* src = emb + (size_t)(cb * 16 + col) * E_DIM + kb;

    f4 x0 = *(const f4*)(src + 0);
    f4 x1 = *(const f4*)(src + 4);
    f4 x2 = *(const f4*)(src + 8);
    f4 x3 = *(const f4*)(src + 12);

    us16 o;
#pragma unroll
    for (int i = 0; i < 4; ++i) {
        o[i]      = f2bf(x0[i]);
        o[4 + i]  = f2bf(x1[i]);
        o[8 + i]  = f2bf(x2[i]);
        o[12 + i] = f2bf(x3[i]);
    }
    *(us16*)(embq + (size_t)(((cb * 8) + ks) * 32 + lane) * 16) = o;
}

// ---------------------------------------------------------------------------
// Kernel 2: ||e_j||^2 table + zero the loss accumulator.  4 blocks x 256.
// ---------------------------------------------------------------------------
__launch_bounds__(256)
__global__ void vq_prep_enorm_k(const float* __restrict__ emb,
                                float* __restrict__ enorm2,
                                float* __restrict__ loss) {
    const int c = blockIdx.x * 256 + threadIdx.x;   // 0..1023
    if (c == 0) *loss = 0.0f;
    const float* er = emb + (size_t)c * E_DIM;
    float s = 0.0f;
#pragma unroll 4
    for (int i = 0; i < E_DIM / 4; ++i) {
        f4 v = ((const f4*)er)[i];
        s += v[0] * v[0] + v[1] * v[1] + v[2] * v[2] + v[3] * v[3];
    }
    enorm2[c] = s;
}

// ---------------------------------------------------------------------------
// Kernel 3: main WMMA argmin + gather + loss.
// 512 workgroups x 256 threads (8 wave32). Wave w owns tokens
// blockIdx*128 + w*16 .. +15; keeps all 8 bf16 A-fragments resident.
// 64 codebook chunks double-buffered through LDS with async loads:
// one s_wait_asynccnt + one barrier per chunk.
// ---------------------------------------------------------------------------
__launch_bounds__(256)
__global__ void vq_main_k(const float* __restrict__ z,
                          const float* __restrict__ emb,
                          const unsigned short* __restrict__ embq,
                          const float* __restrict__ enorm2,
                          float* __restrict__ out,
                          float* __restrict__ loss) {
    __shared__ __align__(16) unsigned short sB[2][8 * 32 * 16]; // 2 x 8KB
    __shared__ __align__(16) float          sE[N_E];            // 4KB ||e||^2
    __shared__ int   sIdx[128];
    __shared__ float sRed[8];

    const int tid   = threadIdx.x;
    const int wave  = tid >> 5;
    const int lane  = tid & 31;
    const int laneN = lane & 15;
    const bool hi   = lane >= 16;
    const int  t0   = blockIdx.x * 128;
    const int  m0   = t0 + wave * 16;

    // LDS byte offsets of the two B buffers (flat LDS address low 32 bits)
    const unsigned sb0 = (unsigned)(size_t)&sB[0][0];
    const unsigned sb1 = (unsigned)(size_t)&sB[1][0];
    const unsigned long long gbase = (unsigned long long)(size_t)embq;

    // stage ||e||^2 once
#pragma unroll
    for (int i = 0; i < N_E / 256; ++i) sE[tid + i * 256] = enorm2[tid + i * 256];

    // kick off async fill of chunk 0 into buffer 0 (32B per thread)
    async_copy32(sb0 + (unsigned)(tid * 16), gbase + (unsigned)(tid * 16));

    // ---- A fragments: bf16, contiguous k order, same as B ----
    // lo lanes: halves = K {b..b+15}, b = ks*32
    // hi lanes: halves = K {b+16..b+31}
    us16 afrag[8];
    {
        const float* zr = z + (size_t)(m0 + laneN) * E_DIM;
#pragma unroll
        for (int ks = 0; ks < 8; ++ks) {
            const int b0 = ks * 32 + (hi ? 16 : 0);
            f4 x0 = *(const f4*)(zr + b0);
            f4 x1 = *(const f4*)(zr + b0 + 4);
            f4 x2 = *(const f4*)(zr + b0 + 8);
            f4 x3 = *(const f4*)(zr + b0 + 12);
            us16 a;
#pragma unroll
            for (int i = 0; i < 4; ++i) {
                a[i]      = f2bf(x0[i]);
                a[4 + i]  = f2bf(x1[i]);
                a[8 + i]  = f2bf(x2[i]);
                a[12 + i] = f2bf(x3[i]);
            }
            afrag[ks] = a;
        }
    }

    float bestd[8];
    int   besti[8];
#pragma unroll
    for (int r = 0; r < 8; ++r) { bestd[r] = 3.4e38f; besti[r] = 0; }

    // chunk 0 must be fully in LDS before anyone computes
    wait_async0();
    __syncthreads();

    for (int cb = 0; cb < 64; ++cb) {
        // issue async fill for next chunk into the other buffer; safe: that
        // buffer's readers finished before the barrier that ended iter cb-1.
        if (cb + 1 < 64) {
            const unsigned dst = (((cb + 1) & 1) ? sb1 : sb0) + (unsigned)(tid * 16);
            async_copy32(dst, gbase + (unsigned long long)(cb + 1) * 8192ull
                              + (unsigned)(tid * 16));
        }
        if (cb + 2 < 64)  // warm L2 two chunks ahead (global_prefetch_b8)
            __builtin_prefetch(embq + (size_t)(cb + 2) * 4096 + lane * 128, 0, 1);

        const unsigned short* B = sB[cb & 1];
        v8f acc = {};
#pragma unroll
        for (int ks = 0; ks < 8; ++ks) {
            us16 braw = *(const us16*)(B + (ks * 32 + lane) * 16);
            acc = __builtin_amdgcn_wmma_f32_16x16x32_bf16(
                false, __builtin_bit_cast(v16bf, afrag[ks]),
                false, __builtin_bit_cast(v16bf, braw),
                (short)0, acc, false, false);
        }

        const int   code = cb * 16 + laneN;
        const float en   = sE[code];
#pragma unroll
        for (int r = 0; r < 8; ++r) {
            float d = en - 2.0f * acc[r];   // C: lane->N, VGPR r -> M (hi: M+8)
            if (d < bestd[r]) { bestd[r] = d; besti[r] = code; }
        }

        // next buffer fully written + all waves done with current buffer
        wait_async0();
        __syncthreads();
    }

    // argmin across the 16 lanes sharing each M-row group (first-index ties)
#pragma unroll
    for (int off = 1; off < 16; off <<= 1) {
#pragma unroll
        for (int r = 0; r < 8; ++r) {
            float od = __shfl_xor(bestd[r], off, 32);
            int   oi = __shfl_xor(besti[r], off, 32);
            if (od < bestd[r] || (od == bestd[r] && oi < besti[r])) {
                bestd[r] = od; besti[r] = oi;
            }
        }
    }
    if (laneN == 0) {
        const int base = wave * 16 + (hi ? 8 : 0);
#pragma unroll
        for (int r = 0; r < 8; ++r) sIdx[base + r] = besti[r];
    }
    __syncthreads();

    // ---- exact fp32 gather + loss: 128 tokens x 256 dims ----
    float lsum = 0.0f;
    const int sub  = tid & 63;  // float4 slot in a 256-wide row
    const int trow = tid >> 6;  // 0..3 tokens per pass
#pragma unroll 4
    for (int p = 0; p < 32; ++p) {
        const int t    = p * 4 + trow;
        const int code = sIdx[t];
        f4 e  = ((const f4*)(emb + (size_t)code * E_DIM))[sub];
        f4 zv = ((const f4*)(z + (size_t)(t0 + t) * E_DIM))[sub];
        ((f4*)(out + (size_t)(t0 + t) * E_DIM))[sub] = e;
        f4 d = zv - e;
        lsum += d[0] * d[0] + d[1] * d[1] + d[2] * d[2] + d[3] * d[3];
    }

#pragma unroll
    for (int off = 16; off >= 1; off >>= 1) lsum += __shfl_xor(lsum, off, 32);
    if (lane == 0) sRed[wave] = lsum;
    __syncthreads();
    if (tid == 0) {
        float s = 0.0f;
#pragma unroll
        for (int w = 0; w < 8; ++w) s += sRed[w];
        // loss = (1+beta) * mean((z - z_q)^2)
        atomicAdd(loss, s * ((1.0f + BETA) / ((float)N_TOK * (float)E_DIM)));
    }
}

// ---------------------------------------------------------------------------
extern "C" void kernel_launch(void* const* d_in, const int* in_sizes, int n_in,
                              void* d_out, int out_size, void* d_ws, size_t ws_size,
                              hipStream_t stream) {
    (void)in_sizes; (void)n_in; (void)out_size; (void)ws_size;
    const float* z   = (const float*)d_in[0];
    const float* emb = (const float*)d_in[1];
    float* out  = (float*)d_out;
    float* loss = out + (size_t)N_TOK * E_DIM;

    // ws: [0, 512KB)  bf16 pre-swizzled codebook; [512KB, +4KB) ||e||^2
    unsigned short* embq = (unsigned short*)d_ws;
    float* enorm2 = (float*)((char*)d_ws + (size_t)N_E * E_DIM * sizeof(unsigned short));

    vq_prep_layout_k<<<64, 256, 0, stream>>>(emb, embq);
    vq_prep_enorm_k<<<4, 256, 0, stream>>>(emb, enorm2, loss);
    vq_main_k<<<N_TOK / 128, 256, 0, stream>>>(z, emb, embq, enorm2, out, loss);
}
